// SparseMoeBlock_6975026889091
// MI455X (gfx1250) — matrix operand order repaired and checked by reading, not compiled, not run on hardware
//
#include <hip/hip_runtime.h>

// MI455X (gfx1250) fused top-2 MoE block.
// Shapes (static): S=1024 B=8 -> T=8192 tokens, D=1024, F=4096, E=8, TOPK=2.
// Route -> bucket tokens per expert -> per 32-token tile run a fused
// (x@w1 -> gelu -> @w2) pipeline on v_wmma_f32_16x16x32_bf16 with fp32->bf16
// conversion on the fly (bf16 == high 16 bits of fp32), double-buffered LDS
// weight staging (1 barrier / K-step), 2 WMMAs per staged B fragment.

#define TD   8192      // tokens
#define DD   1024      // hidden dim
#define FF   4096      // ffn dim
#define FQ   1024      // quarter of FF kept in LDS at a time
#define EE   8         // experts
#define MT   32        // tokens per tile (two WMMA M-subtiles)
#define NTH  512       // 16 waves per block

#define XB_P  1032     // xb row pitch (bf16), +8 pad -> conflict-free ds_load_b128
#define HB_P  1032     // Hb row pitch (bf16)
#define STG_P 40       // staging row pitch (bf16) = 20 dwords
#define STG_DW (256 * STG_P / 2)   // dwords per staging buffer (256 rows)

typedef __attribute__((ext_vector_type(16))) __bf16 v16bf;
typedef __attribute__((ext_vector_type(8)))  __bf16 v8bf;
typedef __attribute__((ext_vector_type(8)))  float  v8f;

__device__ __forceinline__ v8f wmma_bf16(v16bf a, v16bf b, v8f c) {
  // D = A(16x32 bf16) * B(32x16 bf16) + C(16x16 f32)
  return __builtin_amdgcn_wmma_f32_16x16x32_bf16(
      false, a, false, b, (short)0, c, false, false);
}

__device__ __forceinline__ unsigned pack_bf16(float a, float b) {
  return (__float_as_uint(a) >> 16) | (__float_as_uint(b) & 0xFFFF0000u);
}

__device__ __forceinline__ v16bf load_frag(const __bf16* p0, const __bf16* p1) {
  v8bf lo = *(const v8bf*)p0;   // ds_load_b128
  v8bf hi = *(const v8bf*)p1;   // ds_load_b128
  v16bf r;
#pragma unroll
  for (int i = 0; i < 8; ++i) { r[i] = lo[i]; r[i + 8] = hi[i]; }
  return r;
}

// Stage a 32(K) x 256(N) fp32 tile (row stride LD) into an LDS staging buffer
// as bf16, transposed to [n][k] with an 80B pitch (conflict-free b128 reads).
// 512 threads: 16 k-pairs x 32 n-groups of 8.
template <int LD>
__device__ __forceinline__ void stage_tile(const float* __restrict__ g,
                                           unsigned* stg_u32, int tid) {
  const int kp = tid >> 5;          // 0..15 -> rows 2kp, 2kp+1
  const int nb = (tid & 31) << 3;   // 0,8,...,248
  const float* r0 = g + (size_t)(2 * kp) * LD + nb;
  const float* r1 = r0 + LD;
  float4 a0 = *(const float4*)(r0);
  float4 a1 = *(const float4*)(r0 + 4);
  float4 b0 = *(const float4*)(r1);
  float4 b1 = *(const float4*)(r1 + 4);
  __builtin_prefetch(r0 + (size_t)32 * LD, 0, 1);  // next K-slab
  float av[8] = {a0.x, a0.y, a0.z, a0.w, a1.x, a1.y, a1.z, a1.w};
  float bv[8] = {b0.x, b0.y, b0.z, b0.w, b1.x, b1.y, b1.z, b1.w};
#pragma unroll
  for (int j = 0; j < 8; ++j)
    stg_u32[(nb + j) * (STG_P / 2) + kp] = pack_bf16(av[j], bv[j]);
}

// ---------------- Router: one wave32 per token ----------------
__global__ __launch_bounds__(256) void moe_router_kernel(
    const float* __restrict__ x, const float* __restrict__ gw,
    const float* __restrict__ gb, float* __restrict__ logits_out,
    int* __restrict__ counts, int* __restrict__ tok_id,
    float* __restrict__ tok_w) {
  const int lane = threadIdx.x & 31;
  const int t = blockIdx.x * 8 + (threadIdx.x >> 5);

  float acc[EE];
#pragma unroll
  for (int e = 0; e < EE; ++e) acc[e] = 0.f;

  const float* xr = x + (size_t)t * DD;
#pragma unroll 4
  for (int i = 0; i < 32; ++i) {
    const int d = i * 32 + lane;
    const float xv = xr[d];
    float4 g0 = *(const float4*)(gw + d * EE);
    float4 g1 = *(const float4*)(gw + d * EE + 4);
    acc[0] += xv * g0.x; acc[1] += xv * g0.y;
    acc[2] += xv * g0.z; acc[3] += xv * g0.w;
    acc[4] += xv * g1.x; acc[5] += xv * g1.y;
    acc[6] += xv * g1.z; acc[7] += xv * g1.w;
  }
#pragma unroll
  for (int off = 16; off > 0; off >>= 1)
#pragma unroll
    for (int e = 0; e < EE; ++e) acc[e] += __shfl_xor(acc[e], off, 32);

  if (lane == 0) {
#pragma unroll
    for (int e = 0; e < EE; ++e) acc[e] += gb[e];
#pragma unroll
    for (int e = 0; e < EE; ++e) logits_out[(size_t)t * EE + e] = acc[e];
    int i0 = 0; float l0 = acc[0];
#pragma unroll
    for (int e = 1; e < EE; ++e) if (acc[e] > l0) { l0 = acc[e]; i0 = e; }
    int i1 = -1; float l1 = -3.0e38f;
#pragma unroll
    for (int e = 0; e < EE; ++e)
      if (e != i0 && acc[e] > l1) { l1 = acc[e]; i1 = e; }
    // softmax -> top2 -> renorm == 2-way logistic on the two logits
    const float w0 = 1.f / (1.f + __expf(l1 - l0));
    const float w1 = 1.f - w0;
    int p0 = atomicAdd(&counts[i0], 1);
    tok_id[i0 * TD + p0] = t; tok_w[i0 * TD + p0] = w0;
    int p1 = atomicAdd(&counts[i1], 1);
    tok_id[i1 * TD + p1] = t; tok_w[i1 * TD + p1] = w1;
  }
}

// ---------------- Fused expert FFN: one 32-token tile per block ----------------
__global__ __launch_bounds__(NTH) void moe_expert_kernel(
    const float* __restrict__ x, const float* __restrict__ w1,
    const float* __restrict__ b1, const float* __restrict__ w2,
    const float* __restrict__ b2, const int* __restrict__ counts,
    const int* __restrict__ tok_id, const float* __restrict__ tok_w,
    float* __restrict__ out) {
  // ~173 KB (173312 B) LDS of the 320 KB WGP pool.
  __shared__ __align__(16) __bf16 xb[MT * XB_P];        // x tile [32][1024(+8)]
  __shared__ __align__(16) __bf16 Hb[MT * HB_P];        // gelu(h) quarter [32][1024(+8)]
  __shared__ __align__(16) __bf16 stg[2 * 256 * STG_P]; // double-buffered weight slab
  __shared__ int   s_tok[MT];
  __shared__ float s_w[MT];

  const int e = blockIdx.y;
  const int ne = counts[e];
  const int tile0 = blockIdx.x * MT;
  if (tile0 >= ne) return;  // fixed grid for graph capture; idle tiles exit

  const int tid = threadIdx.x;
  const int wv  = tid >> 5;   // wave id 0..15 -> 16-column slice owner
  const int l   = tid & 31;
  const int nl  = l & 15;
  const int hi  = l >> 4;
  unsigned* stg_u32 = (unsigned*)stg;

  if (tid < MT) {
    int idx = tile0 + tid;
    int tk = 0; float wg = 0.f;
    if (idx < ne) { tk = tok_id[e * TD + idx]; wg = tok_w[e * TD + idx]; }
    s_tok[tid] = tk; s_w[tid] = wg;   // pad rows: weight 0 -> no contribution
  }
  __syncthreads();

  {  // gather + convert x rows -> bf16 LDS tile (32 rows, 16 threads/row)
    const int m  = tid >> 4;
    const int c0 = (tid & 15) * 64;
    const float* xr = x + (size_t)s_tok[m] * DD + c0;
    unsigned* xbu = (unsigned*)xb;
    const int base = (m * XB_P + c0) >> 1;
#pragma unroll
    for (int j = 0; j < 16; ++j) {
      float4 v = *(const float4*)(xr + 4 * j);
      xbu[base + 2 * j]     = pack_bf16(v.x, v.y);
      xbu[base + 2 * j + 1] = pack_bf16(v.z, v.w);
    }
  }
  __syncthreads();

  // GEMM2 accumulators: [32 x 1024] tile; wave owns 4 col-tiles x 2 M-subtiles.
  v8f acc2a[4], acc2b[4];
#pragma unroll
  for (int i = 0; i < 4; ++i)
#pragma unroll
    for (int r = 0; r < 8; ++r) { acc2a[i][r] = 0.f; acc2b[i][r] = 0.f; }

  const float* w1e = w1 + (size_t)e * DD * FF;
  const float* w2e = w2 + (size_t)e * FF * DD;

  const __bf16* arow0 = xb + nl * XB_P;          // A subtile 0 (rows 0..15)
  const __bf16* arow1 = xb + (16 + nl) * XB_P;   // A subtile 1 (rows 16..31)
  const __bf16* hrow0 = Hb + nl * HB_P;
  const __bf16* hrow1 = Hb + (16 + nl) * HB_P;
  const int bofs = (wv * 16 + nl) * STG_P + hi * 8;  // B fragment offset (bf16)

  for (int p = 0; p < 4; ++p) {  // four F-quarters so H fits in LDS
    // ---- GEMM1 quarter: Hb = gelu(xb @ w1[:, p*FQ .. p*FQ+FQ] + b1) ----
    for (int ns = 0; ns < 4; ++ns) {       // 4 slabs of 256 F-columns
      v8f acc1a, acc1b;
#pragma unroll
      for (int r = 0; r < 8; ++r) { acc1a[r] = 0.f; acc1b[r] = 0.f; }
      const int fg0 = p * FQ + ns * 256;
      stage_tile<FF>(w1e + fg0, stg_u32, tid);           // ks = 0 -> buf 0
      __syncthreads();
      for (int ks = 0; ks < 32; ++ks) {    // K = D in steps of 32
        const int cur = ks & 1;
        if (ks + 1 < 32)                   // stage next slab into other buffer
          stage_tile<FF>(w1e + (size_t)((ks + 1) * 32) * FF + fg0,
                         stg_u32 + (cur ^ 1) * STG_DW, tid);
        const int kb = ks * 32;
        v16bf a0 = load_frag(arow0 + kb + hi * 8, arow0 + kb + 16 + hi * 8);
        v16bf a1 = load_frag(arow1 + kb + hi * 8, arow1 + kb + 16 + hi * 8);
        const __bf16* bp = stg + cur * (2 * STG_DW) + bofs;
        v16bf b = load_frag(bp, bp + 16);
        acc1a = wmma_bf16(a0, b, acc1a);
        acc1b = wmma_bf16(a1, b, acc1b);
        __syncthreads();                   // one barrier per K-step
      }
      const int fl = ns * 256 + wv * 16 + nl;   // col within this quarter
      const float bias = b1[e * FF + p * FQ + fl];
#pragma unroll
      for (int r = 0; r < 8; ++r) {
        float v0 = acc1a[r] + bias;
        float v1 = acc1b[r] + bias;
        float g0 = 0.5f * v0 * (1.0f + erff(v0 * 0.70710678118f));
        float g1 = 0.5f * v1 * (1.0f + erff(v1 * 0.70710678118f));
        Hb[(r + 8 * hi) * HB_P + fl]      = (__bf16)g0;
        Hb[(16 + r + 8 * hi) * HB_P + fl] = (__bf16)g1;
      }
    }

    // ---- GEMM2 partial: acc2 += Hb @ w2[p*FQ .. p*FQ+FQ, :] ----
#pragma unroll
    for (int ns2 = 0; ns2 < 4; ++ns2) {    // 4 slabs of 256 D-columns
      stage_tile<DD>(w2e + (size_t)(p * FQ) * DD + ns2 * 256, stg_u32, tid);
      __syncthreads();                     // also fences the H-stores above
      for (int ks2 = 0; ks2 < 32; ++ks2) { // K = FQ in steps of 32
        const int cur = ks2 & 1;
        if (ks2 + 1 < 32)
          stage_tile<DD>(w2e + (size_t)(p * FQ + (ks2 + 1) * 32) * DD + ns2 * 256,
                         stg_u32 + (cur ^ 1) * STG_DW, tid);
        const int kb = ks2 * 32;
        v16bf a0 = load_frag(hrow0 + kb + hi * 8, hrow0 + kb + 16 + hi * 8);
        v16bf a1 = load_frag(hrow1 + kb + hi * 8, hrow1 + kb + 16 + hi * 8);
        const __bf16* bp = stg + cur * (2 * STG_DW) + bofs;
        v16bf b = load_frag(bp, bp + 16);
        acc2a[ns2] = wmma_bf16(a0, b, acc2a[ns2]);
        acc2b[ns2] = wmma_bf16(a1, b, acc2b[ns2]);
        __syncthreads();
      }
    }
    __syncthreads();  // Hb fully consumed before next quarter overwrites it
  }

  // ---- bias, routing-weight scale, merge (each token hit by exactly 2 experts)
  float wr0[8], wr1[8]; int tr0[8], tr1[8];
#pragma unroll
  for (int r = 0; r < 8; ++r) {
    wr0[r] = s_w[r + 8 * hi];        tr0[r] = s_tok[r + 8 * hi];
    wr1[r] = s_w[16 + r + 8 * hi];   tr1[r] = s_tok[16 + r + 8 * hi];
  }
#pragma unroll
  for (int ns2 = 0; ns2 < 4; ++ns2) {
    const int col = ns2 * 256 + wv * 16 + nl;
    const float bias = b2[e * DD + col];
#pragma unroll
    for (int r = 0; r < 8; ++r) {
      atomicAdd(&out[(size_t)tr0[r] * DD + col], (acc2a[ns2][r] + bias) * wr0[r]);
      atomicAdd(&out[(size_t)tr1[r] * DD + col], (acc2b[ns2][r] + bias) * wr1[r]);
    }
  }
}

extern "C" void kernel_launch(void* const* d_in, const int* in_sizes, int n_in,
                              void* d_out, int out_size, void* d_ws,
                              size_t ws_size, hipStream_t stream) {
  const float* x  = (const float*)d_in[0];  // [1024,8,1024]
  const float* gw = (const float*)d_in[1];  // [1024,8]
  const float* gb = (const float*)d_in[2];  // [8]
  const float* w1 = (const float*)d_in[3];  // [8,1024,4096]
  const float* b1 = (const float*)d_in[4];  // [8,4096]
  const float* w2 = (const float*)d_in[5];  // [8,4096,1024]
  const float* b2 = (const float*)d_in[6];  // [8,1024]

  float* out    = (float*)d_out;                 // [S,B,D] flat
  float* logits = out + (size_t)TD * DD;         // [T,E] appended

  int*   counts = (int*)d_ws;
  int*   tok_id = (int*)((char*)d_ws + 256);
  float* tok_w  = (float*)((char*)d_ws + 256 + (size_t)EE * TD * 4);

  hipMemsetAsync(d_out, 0, (size_t)TD * DD * sizeof(float), stream);
  hipMemsetAsync(d_ws, 0, 256, stream);

  moe_router_kernel<<<TD / 8, 256, 0, stream>>>(x, gw, gb, logits, counts,
                                                tok_id, tok_w);
  // fixed max grid (worst case: all tokens to one expert); idle tiles exit
  moe_expert_kernel<<<dim3(TD / MT, EE), NTH, 0, stream>>>(
      x, w1, b1, w2, b2, counts, tok_id, tok_w, out);
}
